// BaichuanLayer_43044162240762
// MI455X (gfx1250) — matrix operand order repaired and checked by reading, not compiled, not run on hardware
//
#include <hip/hip_runtime.h>
#include <hip/hip_bf16.h>

// ---------------------------------------------------------------------------
// Baichuan decoder layer for gfx1250 (MI455X), bf16 WMMA everywhere,
// async global->LDS staging where the layout is copy-through.
// ---------------------------------------------------------------------------

#define SEQ    2048
#define HID    4096
#define NHEADS 32
#define HDIM   128
#define QKV3   12288
#define INTER  11008

typedef __attribute__((ext_vector_type(16))) __bf16        v16bf;
typedef __attribute__((ext_vector_type(8)))  float         v8f;
typedef __attribute__((ext_vector_type(4)))  unsigned int  u32x4;

union FragU  { u32x4 u[2]; v16bf v; };            // 32B A/B fragment
union Pack8  { u32x4 u;    unsigned short s[8]; };// 8 bf16 in 16B

// ---- gfx1250 async global->LDS path (guarded; falls back to sync copy) ----
#if defined(__has_builtin)
#  if __has_builtin(__builtin_amdgcn_global_load_async_to_lds_b128)
#    define ASYNC_LDS 1
#  else
#    define ASYNC_LDS 0
#  endif
#else
#  define ASYNC_LDS 0
#endif

// Builtin signature (from clang diagnostic): param0 = AS1 'int __vector(4)*',
// param1 = AS3 counterpart, then two constant ints (imm offset, cpol).
typedef int v4i_g __attribute__((vector_size(16)));
typedef __attribute__((address_space(1))) v4i_g* gas_v4i;
typedef __attribute__((address_space(3))) v4i_g* las_v4i;

__device__ __forceinline__ void stage16(const void* g, void* l) {
#if ASYNC_LDS
    // flat VA of global memory == AS1 VA; LDS aperture: addr[31:0] is LDS offset
    __builtin_amdgcn_global_load_async_to_lds_b128(
        (gas_v4i)(unsigned long long)g,
        (las_v4i)(unsigned int)(unsigned long long)l,
        0, 0);
#else
    *(u32x4*)l = *(const u32x4*)g;
#endif
}

__device__ __forceinline__ void stage_wait() {
#if ASYNC_LDS
#  if __has_builtin(__builtin_amdgcn_s_wait_asynccnt)
    __builtin_amdgcn_s_wait_asynccnt(0);
#  else
    asm volatile("s_wait_asynccnt 0x0" ::: "memory");
#  endif
#endif
}

__device__ __forceinline__ unsigned short f32_to_bf16(float f) {
    union { float f; unsigned u; } v; v.f = f;
    unsigned r = v.u + 0x7FFFu + ((v.u >> 16) & 1u);   // round-to-nearest-even
    return (unsigned short)(r >> 16);
}
__device__ __forceinline__ float bf16_to_f32(unsigned short h) {
    union { unsigned u; float f; } v; v.u = ((unsigned)h) << 16;
    return v.f;
}

// ---------------------------------------------------------------------------
// fp32 -> bf16 streaming cast (weights)
// ---------------------------------------------------------------------------
__global__ __launch_bounds__(256) void cvt_f32_bf16(const float* __restrict__ in,
                                                    unsigned short* __restrict__ out,
                                                    size_t n) {
    size_t i      = (size_t)blockIdx.x * blockDim.x + threadIdx.x;
    size_t stride = (size_t)gridDim.x * blockDim.x;
    for (; i < n; i += stride) out[i] = f32_to_bf16(in[i]);
}

// ---------------------------------------------------------------------------
// RMSNorm: fp32 [rows, HID] -> bf16 [rows, HID]
// ---------------------------------------------------------------------------
__global__ __launch_bounds__(256) void rmsnorm_bf16(const float* __restrict__ x,
                                                    const float* __restrict__ w,
                                                    unsigned short* __restrict__ y) {
    const int row = blockIdx.x;
    const float* xr = x + (size_t)row * HID;
    float ss = 0.f;
    for (int c = threadIdx.x; c < HID; c += 256) { float v = xr[c]; ss += v * v; }
    __shared__ float red[256];
    red[threadIdx.x] = ss;
    __syncthreads();
    for (int s = 128; s > 0; s >>= 1) {
        if (threadIdx.x < s) red[threadIdx.x] += red[threadIdx.x + s];
        __syncthreads();
    }
    const float rs = rsqrtf(red[0] * (1.0f / HID) + 1e-6f);
    unsigned short* yr = y + (size_t)row * HID;
    for (int c = threadIdx.x; c < HID; c += 256)
        yr[c] = f32_to_bf16(xr[c] * rs * w[c]);
}

// ---------------------------------------------------------------------------
// Tiled bf16 WMMA GEMM: C[M,N] = A[M,K] * B[K,N]  (+ epilogue)
//   EPI 0: store bf16
//   EPI 1: C.f32 = res.f32 + acc         (residual add)
//   EPI 2: C.bf16 = silu(aux) * acc      (SwiGLU combine; aux bf16 [M,N])
// Block: 256 threads = 8 waves; tile 128x128x32; wave tile 32x64.
// A tile staged via async global->LDS; B tile transposed on store (sync).
// ---------------------------------------------------------------------------
#define BM 128
#define BN 128
#define BK 32

template <int EPI>
__global__ __launch_bounds__(256)
void gemm_bf16(const unsigned short* __restrict__ A,
               const unsigned short* __restrict__ B,
               void* __restrict__ C,
               const float* __restrict__ res,
               const unsigned short* __restrict__ aux,
               int M, int N, int K) {
    __shared__ unsigned short As[2][BM][BK];   // row-major, 8KB per buffer
    __shared__ unsigned short Bs[2][BN][BK];   // transposed [n][k], 8KB per buffer

    const int tid  = threadIdx.x;
    const int lane = tid & 31;
    const int wave = tid >> 5;        // 0..7
    const int wm   = wave >> 1;       // 0..3 (row group, 32 rows each)
    const int wn   = wave & 1;        // 0..1 (col group, 64 cols each)
    const int m0   = blockIdx.y * BM;
    const int n0   = blockIdx.x * BN;
    const int lh   = lane >> 4;       // half-wave select
    const int ll   = lane & 15;

    v8f acc[2][4];
#pragma unroll
    for (int i = 0; i < 2; ++i)
#pragma unroll
        for (int j = 0; j < 4; ++j) acc[i][j] = (v8f){0,0,0,0,0,0,0,0};

    const int KT = K / BK;

    auto load_tiles = [&](int kt, int buf) {
        const int k0 = kt * BK;
        // A tile: 128x32 bf16 = 512 x 16B chunks, 2 per thread (async -> LDS)
#pragma unroll
        for (int i = 0; i < 2; ++i) {
            int idx = tid + i * 256;
            int r = idx >> 2;
            int c = (idx & 3) * 8;
            stage16(A + (size_t)(m0 + r) * K + k0 + c, &As[buf][r][c]);
        }
        // B tile: 32x128 from global, store transposed into Bs[n][k] (sync)
#pragma unroll
        for (int i = 0; i < 2; ++i) {
            int idx = tid + i * 256;
            int r = idx >> 4;          // k row 0..31
            int c = (idx & 15) * 8;    // n col
            Pack8 d;
            d.u = *(const u32x4*)(B + (size_t)(k0 + r) * N + n0 + c);
#pragma unroll
            for (int j = 0; j < 8; ++j) Bs[buf][c + j][r] = d.s[j];
        }
    };

    load_tiles(0, 0);
    stage_wait();
    __syncthreads();

    for (int kt = 0; kt < KT; ++kt) {
        const int buf = kt & 1;
        if (kt + 1 < KT) load_tiles(kt + 1, buf ^ 1);   // overlaps compute below

        v16bf af[2], bfc[4];
#pragma unroll
        for (int i = 0; i < 2; ++i) {
            // 16-bit A layout: lanes 0-15 hold K 0..7 & 16..23, lanes 16-31 hold 8..15 & 24..31
            const unsigned short* p = &As[buf][wm * 32 + i * 16 + ll][lh * 8];
            FragU f;
            f.u[0] = *(const u32x4*)(p);
            f.u[1] = *(const u32x4*)(p + 16);
            af[i] = f.v;
        }
#pragma unroll
        for (int j = 0; j < 4; ++j) {
            // 16-bit B layout: lane = column; lanes 0-15 K 0..7 & 16..23, lanes 16-31 K 8..15 & 24..31
            const unsigned short* p = &Bs[buf][wn * 64 + j * 16 + ll][lh * 8];
            FragU f;
            f.u[0] = *(const u32x4*)(p);
            f.u[1] = *(const u32x4*)(p + 16);
            bfc[j] = f.v;
        }
#pragma unroll
        for (int i = 0; i < 2; ++i)
#pragma unroll
            for (int j = 0; j < 4; ++j)
                acc[i][j] = __builtin_amdgcn_wmma_f32_16x16x32_bf16(
                    false, af[i], false, bfc[j], (short)0, acc[i][j], false, false);

        stage_wait();       // next tile's async copies have landed
        __syncthreads();
    }

    // Epilogue. C/D layout: VGPR r -> row r (lanes 0-15) / r+8 (lanes 16-31); lane -> column.
#pragma unroll
    for (int i = 0; i < 2; ++i) {
#pragma unroll
        for (int j = 0; j < 4; ++j) {
#pragma unroll
            for (int r = 0; r < 8; ++r) {
                const int row = m0 + wm * 32 + i * 16 + r + 8 * lh;
                const int col = n0 + wn * 64 + j * 16 + ll;
                const size_t idx = (size_t)row * N + col;
                float v = acc[i][j][r];
                if (EPI == 0) {
                    ((unsigned short*)C)[idx] = f32_to_bf16(v);
                } else if (EPI == 1) {
                    ((float*)C)[idx] = res[idx] + v;
                } else { // EPI == 2
                    float g = bf16_to_f32(aux[idx]);
                    float s = g / (1.0f + __expf(-g));
                    ((unsigned short*)C)[idx] = f32_to_bf16(s * v);
                }
            }
        }
    }
}

// ---------------------------------------------------------------------------
// Flash attention (causal), bf16 WMMA, fp32 online softmax.
// qkv: [SEQ, 12288] bf16  (q | k | v, each [SEQ, HID], head-major cols)
// out: [SEQ, HID] bf16
// Block: 256 threads = 8 waves; Q tile 128 rows (16 per wave); KV tile 32.
// Q/K staged async; V transposed on store (sync).
// ---------------------------------------------------------------------------
#define BQ  128
#define BKV 32

__global__ __launch_bounds__(256)
void attn_kernel(const unsigned short* __restrict__ qkv,
                 unsigned short* __restrict__ out) {
    __shared__ unsigned short Qs[BQ][HDIM];       // 32 KB
    __shared__ unsigned short Ks[BKV][HDIM];      //  8 KB (row-major = B layout for Q@K^T)
    __shared__ unsigned short Vs[HDIM][BKV];      //  8 KB (transposed = B layout for P@V)
    __shared__ unsigned short Ps[8][16][BKV];     //  8 KB (per-wave P tile)

    const int head = blockIdx.x;
    const int q0   = blockIdx.y * BQ;
    const int tid  = threadIdx.x;
    const int lane = tid & 31;
    const int wave = tid >> 5;
    const int lh   = lane >> 4;
    const int ll   = lane & 15;
    const size_t qoff = (size_t)head * HDIM;
    const size_t koff = qoff + HID;
    const size_t voff = qoff + 2 * HID;

    // Stage Q tile asynchronously (reused for all KV tiles)
#pragma unroll
    for (int i = 0; i < 8; ++i) {
        int idx = tid + i * 256;
        int r = idx >> 4;
        int c = (idx & 15) * 8;
        stage16(qkv + (size_t)(q0 + r) * QKV3 + qoff + c, &Qs[r][c]);
    }

    float m_i[8], l_i[8];
    v8f o[8];
#pragma unroll
    for (int r = 0; r < 8; ++r) { m_i[r] = -1e30f; l_i[r] = 0.f; }
#pragma unroll
    for (int j = 0; j < 8; ++j) o[j] = (v8f){0,0,0,0,0,0,0,0};

    const float scale = 0.08838834764831845f;   // 1/sqrt(128)
    const int kv_end = q0 + BQ;                 // causal upper bound

    for (int kv0 = 0; kv0 < kv_end; kv0 += BKV) {
        __syncthreads();   // all waves done reading previous K/V tiles
        // Stage K tile 32x128 async; V tile transposed 128x32 sync
#pragma unroll
        for (int i = 0; i < 2; ++i) {
            int idx = tid + i * 256;
            int r = idx >> 4;
            int c = (idx & 15) * 8;
            stage16(qkv + (size_t)(kv0 + r) * QKV3 + koff + c, &Ks[r][c]);
            Pack8 d;
            d.u = *(const u32x4*)(qkv + (size_t)(kv0 + r) * QKV3 + voff + c);
#pragma unroll
            for (int j = 0; j < 8; ++j) Vs[c + j][r] = d.s[j];
        }
        stage_wait();      // Q (first iter) + K async copies landed
        __syncthreads();

        // ---- scores S = Q_tile(16x128) @ K^T(128x32) ----
        v8f s[2];
        s[0] = (v8f){0,0,0,0,0,0,0,0};
        s[1] = (v8f){0,0,0,0,0,0,0,0};
#pragma unroll
        for (int dk = 0; dk < 4; ++dk) {
            const unsigned short* pa = &Qs[wave * 16 + ll][dk * 32 + lh * 8];
            FragU fa;
            fa.u[0] = *(const u32x4*)(pa);
            fa.u[1] = *(const u32x4*)(pa + 16);
#pragma unroll
            for (int t = 0; t < 2; ++t) {
                const unsigned short* pb = &Ks[t * 16 + ll][dk * 32 + lh * 8];
                FragU fb;
                fb.u[0] = *(const u32x4*)(pb);
                fb.u[1] = *(const u32x4*)(pb + 16);
                s[t] = __builtin_amdgcn_wmma_f32_16x16x32_bf16(
                    false, fa.v, false, fb.v, (short)0, s[t], false, false);
            }
        }

        // ---- scale + causal mask ----
        const int qrow_base = q0 + wave * 16 + 8 * lh;
#pragma unroll
        for (int t = 0; t < 2; ++t) {
            const int jcol = kv0 + t * 16 + ll;
#pragma unroll
            for (int r = 0; r < 8; ++r) {
                float v = s[t][r] * scale;
                if (jcol > qrow_base + r) v = -3.0e38f;
                s[t][r] = v;
            }
        }

        // ---- online softmax (row halves live in lane halves; xor<16 stays in half) ----
        float pv[2][8];
        float corr[8];
#pragma unroll
        for (int r = 0; r < 8; ++r) {
            float v = fmaxf(s[0][r], s[1][r]);
            v = fmaxf(v, __shfl_xor(v, 1, 32));
            v = fmaxf(v, __shfl_xor(v, 2, 32));
            v = fmaxf(v, __shfl_xor(v, 4, 32));
            v = fmaxf(v, __shfl_xor(v, 8, 32));
            float mnew = fmaxf(m_i[r], v);
            corr[r] = __expf(m_i[r] - mnew);
            m_i[r] = mnew;
            pv[0][r] = __expf(s[0][r] - mnew);
            pv[1][r] = __expf(s[1][r] - mnew);
            float sum = pv[0][r] + pv[1][r];
            sum += __shfl_xor(sum, 1, 32);
            sum += __shfl_xor(sum, 2, 32);
            sum += __shfl_xor(sum, 4, 32);
            sum += __shfl_xor(sum, 8, 32);
            l_i[r] = l_i[r] * corr[r] + sum;
        }
#pragma unroll
        for (int j = 0; j < 8; ++j)
#pragma unroll
            for (int r = 0; r < 8; ++r) o[j][r] *= corr[r];

        // ---- re-layout P: C-layout -> A-fragment via per-wave LDS ----
#pragma unroll
        for (int t = 0; t < 2; ++t)
#pragma unroll
            for (int r = 0; r < 8; ++r)
                Ps[wave][r + 8 * lh][t * 16 + ll] = f32_to_bf16(pv[t][r]);

        const unsigned short* pp = &Ps[wave][ll][lh * 8];
        FragU fp;
        fp.u[0] = *(const u32x4*)(pp);
        fp.u[1] = *(const u32x4*)(pp + 16);

        // ---- O += P(16x32) @ V(32x128) ----
#pragma unroll
        for (int j = 0; j < 8; ++j) {
            const unsigned short* pb = &Vs[j * 16 + ll][lh * 8];
            FragU fb;
            fb.u[0] = *(const u32x4*)(pb);
            fb.u[1] = *(const u32x4*)(pb + 16);
            o[j] = __builtin_amdgcn_wmma_f32_16x16x32_bf16(
                false, fp.v, false, fb.v, (short)0, o[j], false, false);
        }
    }

    // ---- normalize + store ----
#pragma unroll
    for (int j = 0; j < 8; ++j) {
#pragma unroll
        for (int r = 0; r < 8; ++r) {
            const int row = q0 + wave * 16 + r + 8 * lh;
            const int col = head * HDIM + j * 16 + ll;
            out[(size_t)row * HID + col] = f32_to_bf16(o[j][r] / l_i[r]);
        }
    }
}

// ---------------------------------------------------------------------------
// Launch
// ---------------------------------------------------------------------------
extern "C" void kernel_launch(void* const* d_in, const int* in_sizes, int n_in,
                              void* d_out, int out_size, void* d_ws, size_t ws_size,
                              hipStream_t stream) {
    const float* hidden  = (const float*)d_in[0];   // [1,2048,4096]
    /* d_in[1] = attention_mask: additive causal mask, implemented analytically */
    const float* W_pack  = (const float*)d_in[2];   // [4096,12288]
    const float* o_projW = (const float*)d_in[3];   // [4096,4096]
    const float* gate_w  = (const float*)d_in[4];   // [4096,11008]
    const float* up_w    = (const float*)d_in[5];   // [4096,11008]
    const float* down_w  = (const float*)d_in[6];   // [11008,4096]
    const float* ln1_w   = (const float*)d_in[7];
    const float* ln2_w   = (const float*)d_in[8];
    float* out = (float*)d_out;

    char* ws = (char*)d_ws;
    size_t off = 0;
    auto alloc = [&](size_t bytes) -> void* {
        void* p = ws + off;
        off += (bytes + 255) & ~(size_t)255;
        return p;
    };

    unsigned short* xn1   = (unsigned short*)alloc((size_t)SEQ * HID * 2);
    unsigned short* wpk   = (unsigned short*)alloc((size_t)HID * QKV3 * 2);
    unsigned short* qkv   = (unsigned short*)alloc((size_t)SEQ * QKV3 * 2);
    unsigned short* attnb = (unsigned short*)alloc((size_t)SEQ * HID * 2);
    unsigned short* opw   = (unsigned short*)alloc((size_t)HID * HID * 2);
    float*          hbuf  = (float*)alloc((size_t)SEQ * HID * 4);
    unsigned short* xn2   = (unsigned short*)alloc((size_t)SEQ * HID * 2);
    unsigned short* gw    = (unsigned short*)alloc((size_t)HID * INTER * 2);
    unsigned short* uw    = (unsigned short*)alloc((size_t)HID * INTER * 2);
    unsigned short* dw    = (unsigned short*)alloc((size_t)INTER * HID * 2);
    unsigned short* gbuf  = (unsigned short*)alloc((size_t)SEQ * INTER * 2);
    unsigned short* gact  = (unsigned short*)alloc((size_t)SEQ * INTER * 2);

    // ---- weight casts fp32 -> bf16 ----
    cvt_f32_bf16<<<4096, 256, 0, stream>>>(W_pack,  wpk, (size_t)HID * QKV3);
    cvt_f32_bf16<<<4096, 256, 0, stream>>>(o_projW, opw, (size_t)HID * HID);
    cvt_f32_bf16<<<4096, 256, 0, stream>>>(gate_w,  gw,  (size_t)HID * INTER);
    cvt_f32_bf16<<<4096, 256, 0, stream>>>(up_w,    uw,  (size_t)HID * INTER);
    cvt_f32_bf16<<<4096, 256, 0, stream>>>(down_w,  dw,  (size_t)INTER * HID);

    // ---- attention block ----
    rmsnorm_bf16<<<SEQ, 256, 0, stream>>>(hidden, ln1_w, xn1);

    gemm_bf16<0><<<dim3(QKV3 / BN, SEQ / BM), 256, 0, stream>>>(
        xn1, wpk, qkv, nullptr, nullptr, SEQ, QKV3, HID);

    attn_kernel<<<dim3(NHEADS, SEQ / BQ), 256, 0, stream>>>(qkv, attnb);

    gemm_bf16<1><<<dim3(HID / BN, SEQ / BM), 256, 0, stream>>>(
        attnb, opw, hbuf, hidden, nullptr, SEQ, HID, HID);

    // ---- MLP block ----
    rmsnorm_bf16<<<SEQ, 256, 0, stream>>>(hbuf, ln2_w, xn2);

    gemm_bf16<0><<<dim3(INTER / BN, SEQ / BM), 256, 0, stream>>>(
        xn2, gw, gbuf, nullptr, nullptr, SEQ, INTER, HID);

    gemm_bf16<2><<<dim3(INTER / BN, SEQ / BM), 256, 0, stream>>>(
        xn2, uw, gact, nullptr, gbuf, SEQ, INTER, HID);

    gemm_bf16<1><<<dim3(HID / BN, SEQ / BM), 256, 0, stream>>>(
        gact, dw, out, hbuf, nullptr, SEQ, HID, INTER);

    (void)in_sizes; (void)n_in; (void)out_size; (void)ws_size;
}
